// ResNet_81552839017088
// MI455X (gfx1250) — matrix operand order repaired and checked by reading, not compiled, not run on hardware
//
#include <hip/hip_runtime.h>

// ---------------------------------------------------------------------------
// GATv2 ResNet forward for MI455X (gfx1250, wave32).
// GEMMs: LDS-staged tiles + V_WMMA_F32_16X16X32_F16 (f16 in, f32 accum).
// Block = 8 waves computing a 128x64 tile: 32 WMMA per 32-deep K step.
// Staging uses float4 (global_load_b128) + packed f16 LDS stores.
// Edge phase: float4 gathers + native f32 global atomics (L2-resident).
// ---------------------------------------------------------------------------

typedef __attribute__((ext_vector_type(16))) _Float16 v16h;
typedef __attribute__((ext_vector_type(8)))  _Float16 v8h;
typedef __attribute__((ext_vector_type(4)))  _Float16 v4h;
typedef __attribute__((ext_vector_type(8)))  float    v8f;

#define EPS 1e-5f
#define NEG_SLOPE 0.2f

#define BM 128          // rows per block tile
#define BN 64           // cols per block tile
#define BK 32           // K per step (one WMMA depth)
#define LPAD 40         // LDS row pitch in halves (80B: 16B-aligned, conflict-free)

// ---------------- WMMA GEMM: C[M,Ncols] = A[M,K] @ W[K,Ncols] (+bias)(+relu)
// grid.x = ceil(M/BM), grid.y = Ncols/BN, block = 256 (8 waves).
__global__ __launch_bounds__(256)
void wmma_gemm(const float* __restrict__ A, const float* __restrict__ W,
               const float* __restrict__ bias, float* __restrict__ C,
               int M, int K, int Ncols, int relu) {
  __shared__ _Float16 As[BM * LPAD];          // [row][k]   pitch LPAD
  __shared__ _Float16 Bs[BN * LPAD];          // [col][k]   (transposed on stage)

  const int tid  = threadIdx.x;
  const int wave = tid >> 5;
  const int lane = tid & 31;
  const int m    = lane & 15;                 // row in strip / col in tile
  const int half = lane >> 4;
  const int row0 = blockIdx.x * BM;
  const int col0 = blockIdx.y * BN;
  const int r0   = wave * 16;                 // this wave's row strip

  v8f acc[4] = {{}, {}, {}, {}};

  for (int k0 = 0; k0 < K; k0 += BK) {
    // ---- stage A tile (BM x BK): float4 global loads, b64 f16 LDS stores
    for (int idx = tid; idx < BM * BK / 4; idx += 256) {
      int r = idx >> 3, kk = (idx & 7) << 2;  // 8 float4 per 32-wide row
      int gr = row0 + r; if (gr >= M) gr = M - 1;
      float4 v = *(const float4*)&A[(size_t)gr * K + k0 + kk];
      v4h hv; hv[0] = (_Float16)v.x; hv[1] = (_Float16)v.y;
              hv[2] = (_Float16)v.z; hv[3] = (_Float16)v.w;
      *(v4h*)&As[r * LPAD + kk] = hv;         // 8B-aligned (kk mult of 4)
    }
    // ---- stage B tile (BK x BN) transposed -> Bs[col][k]; b128 global loads
    for (int idx = tid; idx < BK * BN / 4; idx += 256) {
      int k = idx >> 4, c = (idx & 15) << 2;  // 16 float4 per 64-wide row
      float4 v = *(const float4*)&W[(size_t)(k0 + k) * Ncols + col0 + c];
      Bs[(c + 0) * LPAD + k] = (_Float16)v.x;
      Bs[(c + 1) * LPAD + k] = (_Float16)v.y;
      Bs[(c + 2) * LPAD + k] = (_Float16)v.z;
      Bs[(c + 3) * LPAD + k] = (_Float16)v.w;
    }
    // ---- prefetch next K panel into L2 (global_prefetch_b8)
    if (k0 + BK < K) {
      int gr = row0 + (tid >> 1); if (gr >= M) gr = M - 1;
      __builtin_prefetch(&A[(size_t)gr * K + k0 + BK + (tid & 1) * 16], 0, 1);
      if (tid < BK) __builtin_prefetch(&W[(size_t)(k0 + BK + tid) * Ncols + col0], 0, 1);
    }
    __syncthreads();

    // ---- A fragment: lanes 0-15 hold K{0-7,16-23}, lanes 16-31 K{8-15,24-31}
    const _Float16* ap = &As[(r0 + m) * LPAD + half * 8];
    v8h a_lo = *(const v8h*)(ap);             // K = ksel .. ksel+7
    v8h a_hi = *(const v8h*)(ap + 16);        // K = 16+ksel .. 16+ksel+7
    v16h a = __builtin_shufflevector(a_lo, a_hi, 0, 1, 2, 3, 4, 5, 6, 7,
                                     8, 9, 10, 11, 12, 13, 14, 15);
#pragma unroll
    for (int ct = 0; ct < 4; ++ct) {
      // ---- B fragment: col = ct*16+m, K = half*8 .. +7 and 16+half*8 .. +7
      const _Float16* bp = &Bs[(ct * 16 + m) * LPAD + half * 8];
      v8h b_lo = *(const v8h*)(bp);
      v8h b_hi = *(const v8h*)(bp + 16);
      v16h b = __builtin_shufflevector(b_lo, b_hi, 0, 1, 2, 3, 4, 5, 6, 7,
                                       8, 9, 10, 11, 12, 13, 14, 15);
      acc[ct] = __builtin_amdgcn_wmma_f32_16x16x32_f16(
          false, a, false, b, (short)0, acc[ct], false, false);
    }
    __syncthreads();
  }

  // ---- epilogue: D layout VGPR r -> rows r / r+8, col = lane&15
#pragma unroll
  for (int ct = 0; ct < 4; ++ct) {
    int col = col0 + ct * 16 + m;
    float bv = bias ? bias[col] : 0.f;
#pragma unroll
    for (int r = 0; r < 8; ++r) {
      int row = row0 + r0 + r + half * 8;
      if (row < M) {
        float v = acc[ct][r] + bv;
        if (relu) v = v > 0.f ? v : 0.f;
        C[(size_t)row * Ncols + col] = v;
      }
    }
  }
}

// ---------------- small dense layer (64->2 / 64->1 heads, not tile shaped)
__global__ void small_lin(const float* __restrict__ x, const float* __restrict__ W,
                          const float* __restrict__ b, float* __restrict__ out,
                          int N, int K, int O) {
  int t = blockIdx.x * blockDim.x + threadIdx.x;
  if (t >= N * O) return;
  int n = t / O, o = t - n * O;
  float d = b[o];
  for (int k = 0; k < K; ++k) d += x[(size_t)n * K + k] * W[(size_t)k * O + o];
  out[t] = d;
}

// ---------------- elementwise / init kernels
__global__ void add_vec(const float* __restrict__ a, const float* __restrict__ b,
                        float* __restrict__ o, int n) {
  int i = blockIdx.x * blockDim.x + threadIdx.x;
  if (i < n) o[i] = a[i] + b[i];
}

__global__ void fill_bias(float* __restrict__ gout, const float* __restrict__ bias,
                          int n, int HC) {
  int i = blockIdx.x * blockDim.x + threadIdx.x;
  if (i < n) gout[i] = bias[i % HC];
}

__global__ void fill_smax_den(float* __restrict__ smax, float* __restrict__ den, int n) {
  int i = blockIdx.x * blockDim.x + threadIdx.x;
  if (i < n) { smax[i] = -__builtin_inff(); den[i] = 0.f; }
}

__global__ void zero1(float* p) { p[0] = 0.f; }

// ---------------- float atomic max via monotone int/uint mapping
__device__ inline void atomicMaxFloat(float* addr, float v) {
  if (v >= 0.f) atomicMax((int*)addr, __float_as_int(v));
  else          atomicMin((unsigned int*)addr, __float_as_uint(v));
}

// ---------------- GATv2 edge kernels; edge id e >= E is self-loop (src=dst=e-E)
__global__ void gat_score(const float* __restrict__ xl, const float* __restrict__ xr,
                          const float* __restrict__ att, const int* __restrict__ ei,
                          int E, int N, int H, int C,
                          float* __restrict__ score, float* __restrict__ smax) {
  int t = blockIdx.x * blockDim.x + threadIdx.x;
  int ET = E + N;
  if (t >= ET * H) return;
  int e = t / H, h = t - e * H;
  int src, dst;
  if (e < E) { src = ei[e]; dst = ei[E + e]; } else { src = dst = e - E; }
  const float4* pl = (const float4*)(xl + (size_t)src * H * C + (size_t)h * C);
  const float4* pr = (const float4*)(xr + (size_t)dst * H * C + (size_t)h * C);
  const float4* pa = (const float4*)(att + (size_t)h * C);
  float s = 0.f;
  for (int c = 0; c < C / 4; ++c) {
    float4 l = pl[c], r = pr[c], a = pa[c];
    float v0 = l.x + r.x, v1 = l.y + r.y, v2 = l.z + r.z, v3 = l.w + r.w;
    v0 = v0 > 0.f ? v0 : NEG_SLOPE * v0;       // leaky_relu
    v1 = v1 > 0.f ? v1 : NEG_SLOPE * v1;
    v2 = v2 > 0.f ? v2 : NEG_SLOPE * v2;
    v3 = v3 > 0.f ? v3 : NEG_SLOPE * v3;
    s += a.x * v0 + a.y * v1 + a.z * v2 + a.w * v3;
  }
  score[t] = s;
  atomicMaxFloat(&smax[(size_t)dst * H + h], s);
}

__global__ void gat_exp(float* __restrict__ score, const float* __restrict__ smax,
                        float* __restrict__ den, const int* __restrict__ ei,
                        int E, int N, int H) {
  int t = blockIdx.x * blockDim.x + threadIdx.x;
  int ET = E + N;
  if (t >= ET * H) return;
  int e = t / H, h = t - e * H;
  int dst = (e < E) ? ei[E + e] : (e - E);
  float ex = expf(score[t] - smax[(size_t)dst * H + h]);
  score[t] = ex;                                // reuse buffer for exp values
  atomicAdd(&den[(size_t)dst * H + h], ex);
}

__global__ void gat_scatter(const float* __restrict__ ex, const float* __restrict__ den,
                            const float* __restrict__ xl, const int* __restrict__ ei,
                            int E, int N, int H, int C, float* __restrict__ gout) {
  int t = blockIdx.x * blockDim.x + threadIdx.x;
  int ET = E + N;
  if (t >= ET * H) return;
  int e = t / H, h = t - e * H;
  int src, dst;
  if (e < E) { src = ei[e]; dst = ei[E + e]; } else { src = dst = e - E; }
  float alpha = ex[t] / den[(size_t)dst * H + h];
  const float4* pl = (const float4*)(xl + (size_t)src * H * C + (size_t)h * C);
  float* po = gout + (size_t)dst * H * C + (size_t)h * C;
  for (int c = 0; c < C / 4; ++c) {
    float4 l = pl[c];
    atomicAdd(&po[4 * c + 0], alpha * l.x);
    atomicAdd(&po[4 * c + 1], alpha * l.y);
    atomicAdd(&po[4 * c + 2], alpha * l.z);
    atomicAdd(&po[4 * c + 3], alpha * l.w);
  }
}

// ---------------- training-mode BatchNorm (biased variance, like torch)
__global__ void bn_stats(const float* __restrict__ t, float* __restrict__ mean,
                         float* __restrict__ var, int N, int Cch) {
  int c = blockIdx.x;
  __shared__ float ssum[256], ssq[256];
  float s = 0.f, q = 0.f;
  for (int i = threadIdx.x; i < N; i += blockDim.x) {
    float v = t[(size_t)i * Cch + c];
    s += v; q += v * v;
  }
  ssum[threadIdx.x] = s; ssq[threadIdx.x] = q; __syncthreads();
  for (int d = 128; d > 0; d >>= 1) {
    if (threadIdx.x < d) { ssum[threadIdx.x] += ssum[threadIdx.x + d];
                           ssq[threadIdx.x]  += ssq[threadIdx.x + d]; }
    __syncthreads();
  }
  if (threadIdx.x == 0) {
    float mm = ssum[0] / N;
    mean[c] = mm;
    var[c]  = ssq[0] / N - mm * mm;
  }
}

__global__ void bn_apply(const float* __restrict__ t, const float* __restrict__ mean,
                         const float* __restrict__ var, const float* __restrict__ g,
                         const float* __restrict__ b, float* __restrict__ o,
                         int N, int Cch, int relu) {
  int i = blockIdx.x * blockDim.x + threadIdx.x;
  if (i >= N * Cch) return;
  int c = i % Cch;
  float v = (t[i] - mean[c]) * rsqrtf(var[c] + EPS) * g[c] + b[c];
  if (relu) v = v > 0.f ? v : 0.f;
  o[i] = v;
}

// ---------------- value head: sum(tanh(x @ Wv + bv)) then /N
__global__ void value_reduce(const float* __restrict__ x, const float* __restrict__ W,
                             const float* __restrict__ b, float* __restrict__ accum,
                             int N, int K) {
  __shared__ float s[256];
  int i = blockIdx.x * blockDim.x + threadIdx.x;
  float v = 0.f;
  if (i < N) {
    float d = b[0];
    for (int k = 0; k < K; ++k) d += x[(size_t)i * K + k] * W[k];
    v = tanhf(d);
  }
  s[threadIdx.x] = v; __syncthreads();
  for (int d2 = 128; d2 > 0; d2 >>= 1) {
    if (threadIdx.x < d2) s[threadIdx.x] += s[threadIdx.x + d2];
    __syncthreads();
  }
  if (threadIdx.x == 0) atomicAdd(accum, s[0]);
}

__global__ void finalize_value(const float* accum, float* out0, int N) {
  out0[0] = accum[0] / (float)N;
}

// ---------------------------------------------------------------------------
extern "C" void kernel_launch(void* const* d_in, const int* in_sizes, int n_in,
                              void* d_out, int out_size, void* d_ws, size_t ws_size,
                              hipStream_t stream) {
  const int Nn = in_sizes[0] / 128;     // 50000 nodes
  const int Eg = in_sizes[1] / 2;       // 800000 edges
  const int ET = Eg + Nn;               // + self loops

  auto F  = [&](int i) { return (const float*)d_in[i]; };
  const int* ei = (const int*)d_in[1];

  // ---- params in setup_inputs() insertion order
  const float* enc_W = F(2);
  const float* enc_b = F(3);
  // blocks: 3 x 13 arrays starting at index 4
  // policy head: 8 arrays at 43; value head: 8 arrays at 51
  const int BLK0 = 4, HP = 43, HV = 51;

  // ---- workspace carve-up (all fp32)
  float* w = (float*)d_ws;
  size_t off = 0;
  auto alloc = [&](size_t n) { float* p = w + off; off += n; return p; };
  float* x     = alloc((size_t)Nn * 64);
  float* tbuf  = alloc((size_t)Nn * 64);
  float* h64   = alloc((size_t)Nn * 64);
  float* xl    = alloc((size_t)Nn * 256);
  float* xr    = alloc((size_t)Nn * 256);
  float* gout  = alloc((size_t)Nn * 256);
  float* ff    = alloc((size_t)Nn * 512);
  float* score = alloc((size_t)ET * 4);
  float* smax  = alloc((size_t)Nn * 4);
  float* den   = alloc((size_t)Nn * 4);
  float* meanb = alloc(64);
  float* varb  = alloc(64);
  float* accum = alloc(1);
  (void)ws_size; (void)n_in; (void)out_size;

  const int TB = 256;
  auto nb = [&](long n) { return (int)((n + TB - 1) / TB); };

  auto gemm = [&](const float* A, const float* W_, const float* bias, float* C_,
                  int K, int Ncols, int relu) {
    dim3 g((Nn + BM - 1) / BM, Ncols / BN);
    wmma_gemm<<<g, 256, 0, stream>>>(A, W_, bias, C_, Nn, K, Ncols, relu);
  };

  auto run_gat = [&](const float* xin, const float* Wl, const float* Wr,
                     const float* att, const float* bias, int H, int C,
                     float* goutb) {
    int HC = H * C;
    gemm(xin, Wl, nullptr, xl, 64, HC, 0);
    gemm(xin, Wr, nullptr, xr, 64, HC, 0);
    fill_bias<<<nb((long)Nn * HC), TB, 0, stream>>>(goutb, bias, Nn * HC, HC);
    fill_smax_den<<<nb((long)Nn * H), TB, 0, stream>>>(smax, den, Nn * H);
    long tot = (long)ET * H;
    gat_score  <<<nb(tot), TB, 0, stream>>>(xl, xr, att, ei, Eg, Nn, H, C, score, smax);
    gat_exp    <<<nb(tot), TB, 0, stream>>>(score, smax, den, ei, Eg, Nn, H);
    gat_scatter<<<nb(tot), TB, 0, stream>>>(score, den, xl, ei, Eg, Nn, H, C, goutb);
  };

  auto run_bn = [&](const float* tin, const float* g_, const float* b_,
                    float* outp, int relu) {
    bn_stats<<<64, TB, 0, stream>>>(tin, meanb, varb, Nn, 64);
    bn_apply<<<nb((long)Nn * 64), TB, 0, stream>>>(tin, meanb, varb, g_, b_,
                                                   outp, Nn, 64, relu);
  };

  // ---- encoder: x = node_attr @ enc_W + enc_b
  gemm(F(0), enc_W, enc_b, x, 128, 64, 0);

  // ---- 3 residual blocks
  for (int blk = 0; blk < 3; ++blk) {
    int p = BLK0 + 13 * blk;
    const float *Wl = F(p + 0), *Wr = F(p + 1), *att = F(p + 2), *bias = F(p + 3);
    const float *lin = F(p + 4), *bn1g = F(p + 5), *bn1b = F(p + 6);
    const float *W1 = F(p + 7), *b1 = F(p + 8), *W2 = F(p + 9), *b2 = F(p + 10);
    const float *bn2g = F(p + 11), *bn2b = F(p + 12);

    run_gat(x, Wl, Wr, att, bias, 4, 64, gout);
    gemm(gout, lin, nullptr, h64, 256, 64, 0);                 // h = gat @ lin
    add_vec<<<nb((long)Nn * 64), TB, 0, stream>>>(x, h64, tbuf, Nn * 64);
    run_bn(tbuf, bn1g, bn1b, x, 0);                            // x = bn1(x+h)

    gemm(x, W1, b1, ff, 64, 512, 1);                           // relu(x@W1+b1)
    gemm(ff, W2, b2, h64, 512, 64, 0);                         // @W2 + b2
    add_vec<<<nb((long)Nn * 64), TB, 0, stream>>>(x, h64, tbuf, Nn * 64);
    run_bn(tbuf, bn2g, bn2b, x, 0);                            // x = bn2(x+h)
  }

  // ---- policy head -> d_out[1 .. 2N]
  {
    const float *Wl = F(HP + 0), *Wr = F(HP + 1), *att = F(HP + 2), *bias = F(HP + 3);
    const float *bng = F(HP + 4), *bnb = F(HP + 5), *lin = F(HP + 6), *lb = F(HP + 7);
    run_gat(x, Wl, Wr, att, bias, 1, 64, gout);
    run_bn(gout, bng, bnb, h64, 1);                            // relu(bn(gat))
    small_lin<<<nb((long)Nn * 2), TB, 0, stream>>>(h64, lin, lb,
                                                   (float*)d_out + 1, Nn, 64, 2);
  }

  // ---- value head -> d_out[0]
  {
    const float *Wl = F(HV + 0), *Wr = F(HV + 1), *att = F(HV + 2), *bias = F(HV + 3);
    const float *bng = F(HV + 4), *bnb = F(HV + 5), *lin = F(HV + 6), *lb = F(HV + 7);
    run_gat(x, Wl, Wr, att, bias, 1, 64, gout);
    run_bn(gout, bng, bnb, h64, 1);
    zero1<<<1, 1, 0, stream>>>(accum);
    value_reduce<<<nb(Nn), TB, 0, stream>>>(h64, lin, lb, accum, Nn, 64);
    finalize_value<<<1, 1, 0, stream>>>(accum, (float*)d_out, Nn);
  }
}
